// MoELoRA_47931835023557
// MI455X (gfx1250) — matrix operand order repaired and checked by reading, compiled, not run on hardware
//
#include <hip/hip_runtime.h>

typedef __bf16 bf16_t;
typedef __attribute__((ext_vector_type(16))) __bf16 v16bf;
typedef __attribute__((ext_vector_type(8)))  float  v8f;
typedef __attribute__((ext_vector_type(4)))  unsigned int u32x4;
typedef int i32x4 __attribute__((vector_size(16)));   // matches builtin param pointee

#define N_TOK   4096
#define D_IN    1024
#define D_OUT   1024
#define NEXP    8
#define RANK    16
#define KAUG    1152      // D_IN + NEXP*RANK
#define TCOLS   256       // padded cols of T: [0,128)=t(e,r), [128,136)=gate logits, rest 0
#define SCALING 0.0625f   // 1/RANK

// ---- CDNA5 async global->LDS path (guarded so the file always compiles) ----
#if defined(__has_builtin)
#if __has_builtin(__builtin_amdgcn_global_load_async_to_lds_b128)
#define USE_ASYNC 1
#endif
#endif
#ifndef USE_ASYNC
#define USE_ASYNC 0
#endif

// global (addrspace 1) / LDS (addrspace 3) typed-pointer casts for the builtin
#define ASG(p) ((__attribute__((address_space(1))) i32x4*)(p))
#define ASL(p) ((__attribute__((address_space(3))) i32x4*)(p))

template <int N> __device__ __forceinline__ void wait_async() {
#if USE_ASYNC
#if __has_builtin(__builtin_amdgcn_s_wait_asynccnt)
    __builtin_amdgcn_s_wait_asynccnt(N);
#else
    asm volatile("s_wait_asynccnt %0" :: "i"(N) : "memory");
#endif
#endif
}

// ---------------- pack kernels ----------------
__global__ __launch_bounds__(256)
void pack_x_kernel(const float* __restrict__ x, bf16_t* __restrict__ A) {
    int idx = blockIdx.x * 256 + threadIdx.x;          // over N_TOK*D_IN
    int n = idx >> 10, h = idx & 1023;
    A[n * KAUG + h] = (bf16_t)x[idx];
}

__global__ __launch_bounds__(256)
void pack_B_kernel(const float* __restrict__ weight, const float* __restrict__ lora_V,
                   bf16_t* __restrict__ Bm) {
    int idx = blockIdx.x * 256 + threadIdx.x;          // over D_OUT*KAUG
    int o = idx / KAUG, k = idx - o * KAUG;
    float v;
    if (k < D_IN) {
        v = weight[o * D_IN + k];
    } else {
        int j = k - D_IN;                              // j = e*RANK + r
        int e = j >> 4, r = j & 15;
        v = lora_V[(e * D_OUT + o) * RANK + r];
    }
    Bm[idx] = (bf16_t)v;
}

__global__ __launch_bounds__(256)
void pack_U_kernel(const float* __restrict__ lora_U, const float* __restrict__ gate_w,
                   bf16_t* __restrict__ Ug) {
    int idx = blockIdx.x * 256 + threadIdx.x;          // over TCOLS*D_IN
    int r = idx >> 10, h = idx & 1023;
    float v = 0.0f;
    if (r < NEXP * RANK)              v = lora_U[r * D_IN + h];     // (E,R,D) flat
    else if (r < NEXP * RANK + NEXP)  v = gate_w[(r - NEXP * RANK) * D_IN + h];
    Ug[idx] = (bf16_t)v;
}

// ---------------- softmax gate + scaled s ----------------
__global__ __launch_bounds__(256)
void gate_s_kernel(const float* __restrict__ T, bf16_t* __restrict__ A) {
    int n = blockIdx.x * 256 + threadIdx.x;            // token id
    const float* tn = T + (long)n * TCOLS;
    float logit[NEXP], m = -1e30f;
    #pragma unroll
    for (int e = 0; e < NEXP; ++e) { logit[e] = tn[128 + e]; m = fmaxf(m, logit[e]); }
    float g[NEXP], sum = 0.0f;
    #pragma unroll
    for (int e = 0; e < NEXP; ++e) { g[e] = __expf(logit[e] - m); sum += g[e]; }
    float inv = SCALING / sum;
    bf16_t* s = A + (long)n * KAUG + D_IN;
    #pragma unroll 8
    for (int j = 0; j < NEXP * RANK; ++j)
        s[j] = (bf16_t)(g[j >> 4] * inv * tn[j]);
}

// ---------------- bf16 WMMA GEMM: C(MxN) = A(MxK) * B(NxK)^T ----------------
// 128 threads = 4 wave32s; block tile 128x128, BK=32; wave tile 64x64 (4x4 WMMA tiles)
// Double-buffered LDS staging via GLOBAL_LOAD_ASYNC_TO_LDS_B128 (ASYNCcnt pipeline).
#define LDT 40   // padded LDS row in bf16 (80B, 16B-aligned)

struct Frag { union { v16bf v; u32x4 q[2]; }; };

__device__ __forceinline__
void stage_tile(const bf16_t* __restrict__ A, const bf16_t* __restrict__ Bm,
                bf16_t* As, bf16_t* Bs, long rowBase, long colBase,
                int lda, int ldb, int kb, int tid) {
    // 512 chunks of 8 bf16 (16B) per matrix, 128 threads -> 4 chunks each
    #pragma unroll
    for (int i = 0; i < 4; ++i) {
        int c   = i * 128 + tid;
        int row = c >> 2;
        int ko  = (c & 3) * 8;
        const bf16_t* ga = A  + (rowBase + row) * (long)lda + kb + ko;
        const bf16_t* gb = Bm + (colBase + row) * (long)ldb + kb + ko;
#if USE_ASYNC
        __builtin_amdgcn_global_load_async_to_lds_b128(ASG(ga), ASL(As + row * LDT + ko), 0, 0);
        __builtin_amdgcn_global_load_async_to_lds_b128(ASG(gb), ASL(Bs + row * LDT + ko), 0, 0);
#else
        *reinterpret_cast<u32x4*>(As + row * LDT + ko) = *reinterpret_cast<const u32x4*>(ga);
        *reinterpret_cast<u32x4*>(Bs + row * LDT + ko) = *reinterpret_cast<const u32x4*>(gb);
        __builtin_prefetch((const void*)(ga + 32), 0, 1);
#endif
    }
}

__global__ __launch_bounds__(128)
void wmma_gemm_kernel(const bf16_t* __restrict__ A, const bf16_t* __restrict__ Bm,
                      float* __restrict__ C, int K, int lda, int ldb, int ldc) {
    constexpr int BK = 32;
    __shared__ alignas(16) bf16_t As[2][128 * LDT];
    __shared__ alignas(16) bf16_t Bs[2][128 * LDT];

    const int tid  = threadIdx.x;
    const int lane = tid & 31;
    const int wave = tid >> 5;                         // 0..3
    const int l15  = lane & 15;
    const int lh   = lane >> 4;                        // half-wave select
    const int wm   = (wave & 1) * 64;                  // wave row offset in block tile
    const int wn   = (wave >> 1) * 64;                 // wave col offset in block tile
    const long rowBase = (long)blockIdx.y * 128;
    const long colBase = (long)blockIdx.x * 128;

    v8f acc[4][4] = {};

    // prologue: stage first K-slab into buffer 0
    stage_tile(A, Bm, As[0], Bs[0], rowBase, colBase, lda, ldb, 0, tid);

    int buf = 0;
    for (int kb = 0; kb < K; kb += BK, buf ^= 1) {
        if (kb + BK < K) {
            // issue next slab into the other buffer (safe: everyone finished
            // reading it at the previous end-of-iteration barrier)
            stage_tile(A, Bm, As[buf ^ 1], Bs[buf ^ 1], rowBase, colBase,
                       lda, ldb, kb + BK, tid);
            wait_async<8>();       // drain the 8 older copies filling As/Bs[buf]
        } else {
            wait_async<0>();
        }
        __syncthreads();

        // A-frag per ISA layout: lane<16 -> K{0..7,16..23}, lane>=16 -> K{8..15,24..31}
        Frag a[4], b[4];
        #pragma unroll
        for (int i = 0; i < 4; ++i) {
            const bf16_t* p = &As[buf][(wm + i * 16 + l15) * LDT];
            a[i].q[0] = *reinterpret_cast<const u32x4*>(p + lh * 8);
            a[i].q[1] = *reinterpret_cast<const u32x4*>(p + 16 + lh * 8);
        }
        // B-frag per ISA layout: lane<16 -> K{0..7,16..23}, lane>=16 -> K{8..15,24..31}
        #pragma unroll
        for (int j = 0; j < 4; ++j) {
            const bf16_t* p = &Bs[buf][(wn + j * 16 + l15) * LDT];
            b[j].q[0] = *reinterpret_cast<const u32x4*>(p + lh * 8);
            b[j].q[1] = *reinterpret_cast<const u32x4*>(p + 16 + lh * 8);
        }
        #pragma unroll
        for (int i = 0; i < 4; ++i)
            #pragma unroll
            for (int j = 0; j < 4; ++j)
                acc[i][j] = __builtin_amdgcn_wmma_f32_16x16x32_bf16(
                    false, a[i].v, false, b[j].v, (short)0, acc[i][j], false, false);
        __syncthreads();   // buf may be overwritten by async copies next iteration
    }

    // epilogue: C/D layout -> M = g + 8*lh, N = l15
    #pragma unroll
    for (int i = 0; i < 4; ++i)
        #pragma unroll
        for (int j = 0; j < 4; ++j) {
            float* cp = C + (rowBase + wm + i * 16 + 8 * lh) * (long)ldc
                          + colBase + wn + j * 16 + l15;
            #pragma unroll
            for (int g = 0; g < 8; ++g)
                cp[(long)g * ldc] = acc[i][j][g];
        }
}

// ---------------- launch ----------------
extern "C" void kernel_launch(void* const* d_in, const int* in_sizes, int n_in,
                              void* d_out, int out_size, void* d_ws, size_t ws_size,
                              hipStream_t stream) {
    const float* x      = (const float*)d_in[0];
    const float* weight = (const float*)d_in[1];
    const float* gate_w = (const float*)d_in[2];
    const float* lora_U = (const float*)d_in[3];
    const float* lora_V = (const float*)d_in[4];
    float* out = (float*)d_out;

    char* ws = (char*)d_ws;
    size_t offA = 0;
    size_t offB = offA + (size_t)N_TOK * KAUG * sizeof(bf16_t);   //  9.44 MB: A = [x_bf16 | s]
    size_t offU = offB + (size_t)D_OUT * KAUG * sizeof(bf16_t);   // +2.36 MB: B_cat = [W | V]
    size_t offT = offU + (size_t)TCOLS * D_IN * sizeof(bf16_t);   // +0.50 MB: U_g
    bf16_t* Abuf = (bf16_t*)(ws + offA);
    bf16_t* Bcat = (bf16_t*)(ws + offB);
    bf16_t* Ug   = (bf16_t*)(ws + offU);
    float*  T    = (float*)(ws + offT);                           // +4 MB: T = x@Ug^T

    pack_x_kernel<<<(N_TOK * D_IN) / 256, 256, 0, stream>>>(x, Abuf);
    pack_B_kernel<<<(D_OUT * KAUG) / 256, 256, 0, stream>>>(weight, lora_V, Bcat);
    pack_U_kernel<<<(TCOLS * D_IN) / 256, 256, 0, stream>>>(lora_U, gate_w, Ug);

    // T = x_bf @ Ug^T : (4096 x 256), K = 1024  (t and gate logits in one GEMM)
    wmma_gemm_kernel<<<dim3(TCOLS / 128, N_TOK / 128), 128, 0, stream>>>(
        Abuf, Ug, T, D_IN, KAUG, D_IN, TCOLS);

    // softmax gate, s = scaling * gate * t -> A[:, 1024:1152] (bf16)
    gate_s_kernel<<<N_TOK / 256, 256, 0, stream>>>(T, Abuf);

    // out = [x_bf | s] @ [W | V]^T : (4096 x 1024), K = 1152 (base + LoRA fused)
    wmma_gemm_kernel<<<dim3(D_OUT / 128, N_TOK / 128), 128, 0, stream>>>(
        Abuf, Bcat, out, KAUG, KAUG, KAUG, D_OUT);
}
